// MultiHeadSelfAttention_26834955665547
// MI455X (gfx1250) — matrix operand order repaired and checked by reading, not compiled, not run on hardware
//
#include <hip/hip_runtime.h>
#include <stdint.h>

typedef __bf16 bf16;
typedef __attribute__((ext_vector_type(16))) __bf16 v16bf;
typedef __attribute__((ext_vector_type(8)))  float  v8f;

#define HD   1024
#define NH   16
#define DH   64
#define SEQ  2048
#define BATCH 2
#define MTOT (BATCH * SEQ)   // 4096

// padded LDS row strides (odd multiples of 8 elements = 16B, to spread banks)
#define SK_LD 72   // K tile rows of 64 elems, padded
#define SV_LD 40   // V tile rows of 32 elems, padded
#define SG_LD 40   // GEMM tile rows of 32 elems, padded

union FragU { v16bf v; unsigned int u[8]; };
union PackU { unsigned int u; __bf16 e[2]; };

__device__ __forceinline__ v8f wmma_bf16(v16bf a, v16bf b, v8f c) {
  // D = A(16x32 bf16) * B(32x16 bf16) + C(16x16 f32)
  return __builtin_amdgcn_wmma_f32_16x16x32_bf16(false, a, false, b,
                                                 (short)0, c, false, false);
}

__device__ __forceinline__ float lane_read(float v, int srcLane) {
  return __int_as_float(__builtin_amdgcn_ds_bpermute(srcLane << 2, __float_as_int(v)));
}

__device__ __forceinline__ v8f zero_v8f() {
  v8f z;
#pragma unroll
  for (int i = 0; i < 8; ++i) z[i] = 0.0f;
  return z;
}

// Async copy of 16 bytes global -> LDS (per lane). ASYNCcnt-tracked.
__device__ __forceinline__ void async_copy16(void* lds, const void* g) {
  unsigned l = (unsigned)(size_t)lds;                 // LDS byte offset (addr[31:0])
  unsigned long long ga = (unsigned long long)(size_t)g;
  asm volatile("global_load_async_to_lds_b128 %0, %1, off"
               :: "v"(l), "v"(ga) : "memory");
}

__device__ __forceinline__ void wait_async0() {
  asm volatile("s_wait_asynccnt 0x0" ::: "memory");
}

// A fragment (16x32 bf16): rows row0..row0+15, cols k0..k0+31 of row-major src.
// ISA layout: lane(0-15)=rows, VGPR v<4: K=2v+8h ; v>=4: K=16+2(v-4)+8h (h=lane>>4)
__device__ __forceinline__ v16bf load_frag_a(const bf16* src, int ld, int row0, int k0) {
  const int lane = threadIdx.x & 31;
  const int r = lane & 15, h = lane >> 4;
  const bf16* base = src + (size_t)(row0 + r) * ld + (k0 + 8 * h);
  FragU f;
#pragma unroll
  for (int v = 0; v < 4; ++v) f.u[v]     = *(const unsigned int*)(base + 2 * v);
#pragma unroll
  for (int v = 0; v < 4; ++v) f.u[4 + v] = *(const unsigned int*)(base + 16 + 2 * v);
  return f.v;
}

// B fragment (KxN = 32x16 bf16) loaded from B-transposed storage srcT[n][k] (row-major).
// ISA layout: lane holds column n = lane&15, VGPR v<4: K = k0 + 8*(lane>>4) + 2v ; v>=4: K = k0 + 16 + 8*(lane>>4) + 2(v-4) {+0,+1}
__device__ __forceinline__ v16bf load_frag_b(const bf16* srcT, int ld, int n0, int k0) {
  const int lane = threadIdx.x & 31;
  const int n = lane & 15, h = lane >> 4;
  const bf16* base = srcT + (size_t)(n0 + n) * ld + (k0 + 8 * h);
  FragU f;
#pragma unroll
  for (int v = 0; v < 8; ++v) f.u[v] = *(const unsigned int*)(base + (v < 4 ? 2 * v : 16 + 2 * (v - 4)));
  return f.v;
}

__global__ void cvt_f32_bf16(const float* __restrict__ src, bf16* __restrict__ dst, int n4) {
  int i = blockIdx.x * blockDim.x + threadIdx.x;
  if (i < n4) {
    const float4 f = ((const float4*)src)[i];
    union { unsigned long long u; __bf16 e[4]; } p;
    p.e[0] = (bf16)f.x; p.e[1] = (bf16)f.y; p.e[2] = (bf16)f.z; p.e[3] = (bf16)f.w;
    ((unsigned long long*)dst)[i] = p.u;
  }
}

// Stage A[128][32] and Bt[128][32] k-step tiles into padded LDS (4 async b128/thread)
__device__ __forceinline__ void stage_gemm(bf16* sa, bf16* sb,
                                           const bf16* A, const bf16* Bt,
                                           int m0, int n0, int k0) {
  const int t = threadIdx.x;
#pragma unroll
  for (int i = 0; i < 2; ++i) {
    int c = t + i * 256;                  // 0..511 chunks of 16B
    int row = c >> 2, seg = (c & 3) << 3; // 4 chunks per 32-elem row
    async_copy16(sa + row * SG_LD + seg, A  + (size_t)(m0 + row) * HD + k0 + seg);
    async_copy16(sb + row * SG_LD + seg, Bt + (size_t)(n0 + row) * HD + k0 + seg);
  }
}

// out[m,n] = sum_k A[m,k]*Wt[n,k] + bias[n]   (M=MTOT, N=K=HD)
// MODE 0: bf16 row-major out; MODE 1: bf16 vT[b][h][d][m]; MODE 2: f32 row-major out
template <int MODE>
__global__ __launch_bounds__(256)
void gemm_bf16_wmma(const bf16* __restrict__ A, const bf16* __restrict__ Bt,
                    const float* __restrict__ bias, void* __restrict__ out) {
  __shared__ __align__(16) bf16 sa[2][128 * SG_LD];
  __shared__ __align__(16) bf16 sb[2][128 * SG_LD];

  const int wave = threadIdx.x >> 5;
  const int lane = threadIdx.x & 31;
  const int wm = wave & 3, wn = wave >> 2;
  const int m0 = blockIdx.x * 128;
  const int n0 = blockIdx.y * 128;

  v8f acc[2][4];
#pragma unroll
  for (int i = 0; i < 2; ++i)
#pragma unroll
    for (int j = 0; j < 4; ++j) acc[i][j] = zero_v8f();

  int buf = 0;
  stage_gemm(sa[0], sb[0], A, Bt, m0, n0, 0);

  for (int k0 = 0; k0 < HD; k0 += 32) {
    wait_async0();        // my async writes into buf complete
    __syncthreads();      // everyone's writes done; prior reads of buf^1 done
    if (k0 + 32 < HD) stage_gemm(sa[buf ^ 1], sb[buf ^ 1], A, Bt, m0, n0, k0 + 32);

    // all fragments in one ds clause (scheduler fence), then the WMMA chain
    v16bf a0 = load_frag_a(sa[buf], SG_LD, wm * 32,      0);
    v16bf a1 = load_frag_a(sa[buf], SG_LD, wm * 32 + 16, 0);
    v16bf b0 = load_frag_b(sb[buf], SG_LD, wn * 64,      0);
    v16bf b1 = load_frag_b(sb[buf], SG_LD, wn * 64 + 16, 0);
    v16bf b2 = load_frag_b(sb[buf], SG_LD, wn * 64 + 32, 0);
    v16bf b3 = load_frag_b(sb[buf], SG_LD, wn * 64 + 48, 0);
    __builtin_amdgcn_sched_barrier(0);   // keep loads batched: single s_wait_dscnt

    acc[0][0] = wmma_bf16(a0, b0, acc[0][0]);
    acc[1][0] = wmma_bf16(a1, b0, acc[1][0]);
    acc[0][1] = wmma_bf16(a0, b1, acc[0][1]);
    acc[1][1] = wmma_bf16(a1, b1, acc[1][1]);
    acc[0][2] = wmma_bf16(a0, b2, acc[0][2]);
    acc[1][2] = wmma_bf16(a1, b2, acc[1][2]);
    acc[0][3] = wmma_bf16(a0, b3, acc[0][3]);
    acc[1][3] = wmma_bf16(a1, b3, acc[1][3]);
    buf ^= 1;
  }

  const int c = lane & 15, hh = lane >> 4;
  const int wm0 = m0 + wm * 32, wn0 = n0 + wn * 64;
#pragma unroll
  for (int i = 0; i < 2; ++i) {
#pragma unroll
    for (int j = 0; j < 4; ++j) {
      const int n = wn0 + 16 * j + c;
      const float bn = bias[n];
      if (MODE == 0) {
        bf16* o = (bf16*)out;
#pragma unroll
        for (int r = 0; r < 8; ++r) {
          int m = wm0 + 16 * i + 8 * hh + r;
          o[(size_t)m * HD + n] = (bf16)(acc[i][j][r] + bn);
        }
      } else if (MODE == 2) {
        float* o = (float*)out;
#pragma unroll
        for (int r = 0; r < 8; ++r) {
          int m = wm0 + 16 * i + 8 * hh + r;
          o[(size_t)m * HD + n] = acc[i][j][r] + bn;
        }
      } else {
        // vT[(b*16+h)*64 + d][m_local] = (b*HD + n)*SEQ + m_local
        bf16* o = (bf16*)out;
        int mbase = wm0 + 16 * i + 8 * hh;   // 8 consecutive rows, same batch
        int bb = mbase >> 11;
        int ml = mbase & (SEQ - 1);
        bf16* dst = o + ((size_t)bb * HD + n) * SEQ + ml;
#pragma unroll
        for (int r2 = 0; r2 < 4; ++r2) {
          PackU p;
          p.e[0] = (bf16)(acc[i][j][2 * r2]     + bn);
          p.e[1] = (bf16)(acc[i][j][2 * r2 + 1] + bn);
          *(unsigned int*)(dst + 2 * r2) = p.u;
        }
      }
    }
  }
}

// Stage one 32-key step of K ([32 keys][64 d]) and V ([64 d][32 keys]) into LDS.
__device__ __forceinline__ void stage_tiles(bf16* skbuf, bf16* svbuf,
                                            const bf16* khead, const bf16* vhead,
                                            int n0) {
  const int t = threadIdx.x;
  {
    int row = t >> 3, c8 = (t & 7) << 3;
    async_copy16(skbuf + row * SK_LD + c8, khead + (size_t)(n0 + row) * HD + c8);
  }
  {
    int d = t >> 2, s8 = (t & 3) << 3;
    async_copy16(svbuf + d * SV_LD + s8, vhead + (size_t)d * SEQ + n0 + s8);
  }
}

// One wave per 16-query tile; 8 waves per block share one (batch, head) and its
// K/V tiles via double-buffered async LDS staging. Flash softmax on S^T tiles.
__global__ __launch_bounds__(256)
void attention_wmma(const bf16* __restrict__ qb, const bf16* __restrict__ kb,
                    const bf16* __restrict__ vT, bf16* __restrict__ ctx) {
  __shared__ __align__(16) bf16 sk[2][32 * SK_LD];
  __shared__ __align__(16) bf16 sv[2][64 * SV_LD];

  const int wave = threadIdx.x >> 5;
  const int lane = threadIdx.x & 31;
  // block -> (b, h, 128-query slab); wave -> 16-query tile inside the slab
  const int qslab = blockIdx.x & 15;
  const int h     = (blockIdx.x >> 4) & 15;
  const int b     = blockIdx.x >> 8;
  const int q0    = qslab * 128 + wave * 16;

  const bf16* qhead = qb + (size_t)(b * SEQ) * HD + h * DH;
  const bf16* khead = kb + (size_t)(b * SEQ) * HD + h * DH;
  const bf16* vhead = vT + ((size_t)(b * NH + h) * DH) * SEQ;

  // Q fragments (B-operand of S^T wmma), loop invariant
  const v16bf qf0 = load_frag_b(qhead + (size_t)q0 * HD, HD, 0, 0);
  const v16bf qf1 = load_frag_b(qhead + (size_t)q0 * HD, HD, 0, 32);

  v8f o[4];
#pragma unroll
  for (int dt = 0; dt < 4; ++dt) o[dt] = zero_v8f();

  float m_run = -1e30f, l_run = 0.0f;
  const int qg = q0 + (lane & 15);
  const int hh = lane >> 4;

  int buf = 0;
  stage_tiles(sk[0], sv[0], khead, vhead, 0);

  for (int n0 = 0; n0 < SEQ; n0 += 32) {
    wait_async0();        // my async writes into buf are done
    __syncthreads();      // everyone's writes done; prev reads of buf^1 done
    if (n0 + 32 < SEQ) stage_tiles(sk[buf ^ 1], sv[buf ^ 1], khead, vhead, n0 + 32);

    const bf16* skb = sk[buf];
    const bf16* svb = sv[buf];

    // Load ALL fragments for this step in one clause (scheduler fence below):
    // 4 K-frags + 4 V-frags -> single s_wait_dscnt; V latency overlaps softmax.
    v16bf ka00 = load_frag_a(skb, SK_LD, 0,  0);
    v16bf ka01 = load_frag_a(skb, SK_LD, 0,  32);
    v16bf ka10 = load_frag_a(skb, SK_LD, 16, 0);
    v16bf ka11 = load_frag_a(skb, SK_LD, 16, 32);
    v16bf vb0  = load_frag_b(svb + 0 * 16 * SV_LD, SV_LD, 0, 0);
    v16bf vb1  = load_frag_b(svb + 1 * 16 * SV_LD, SV_LD, 0, 0);
    v16bf vb2  = load_frag_b(svb + 2 * 16 * SV_LD, SV_LD, 0, 0);
    v16bf vb3  = load_frag_b(svb + 3 * 16 * SV_LD, SV_LD, 0, 0);
    __builtin_amdgcn_sched_barrier(0);

    // S^T tiles: [16 keys x 16 queries], keys n0..+15 (st0), n0+16..+31 (st1)
    v8f st0 = zero_v8f(), st1 = zero_v8f();
    st0 = wmma_bf16(ka00, qf0, st0);
    st0 = wmma_bf16(ka01, qf1, st0);
    st1 = wmma_bf16(ka10, qf0, st1);
    st1 = wmma_bf16(ka11, qf1, st1);

    // scale by 1/sqrt(64); multiplicative diagonal-zero mask only when the
    // diagonal intersects this 32-key x 16-query window (wave-uniform test)
    float mx = -1e30f;
    const bool diag_hit = (q0 < n0 + 32) && (n0 < q0 + 16);
    if (diag_hit) {
#pragma unroll
      for (int r = 0; r < 8; ++r) {
        int k0g = n0 + r + 8 * hh;
        float s0 = st0[r] * 0.125f;
        float s1 = st1[r] * 0.125f;
        s0 = (k0g == qg)      ? 0.0f : s0;
        s1 = (k0g + 16 == qg) ? 0.0f : s1;
        st0[r] = s0; st1[r] = s1;
        mx = fmaxf(mx, fmaxf(s0, s1));
      }
    } else {
#pragma unroll
      for (int r = 0; r < 8; ++r) {
        float s0 = st0[r] * 0.125f;
        float s1 = st1[r] * 0.125f;
        st0[r] = s0; st1[r] = s1;
        mx = fmaxf(mx, fmaxf(s0, s1));
      }
    }
    // sync max across the lane pair handling the same query (disjoint key halves)
    mx = fmaxf(mx, lane_read(mx, lane ^ 16));
    float m_new = fmaxf(m_run, mx);
    float alpha = __expf(m_run - m_new);
    m_run = m_new;

    float e0[8], e1[8], lsum = 0.0f;
#pragma unroll
    for (int r = 0; r < 8; ++r) {
      e0[r] = __expf(st0[r] - m_new);
      e1[r] = __expf(st1[r] - m_new);
      lsum += e0[r] + e1[r];
    }
    l_run = l_run * alpha + lsum;

    // Repack P^T (D-layout) -> P (A-layout): purely per-lane (layouts align)
    FragU pf;
#pragma unroll
    for (int v = 0; v < 4; ++v) {
      PackU p0; p0.e[0] = (bf16)e0[2 * v]; p0.e[1] = (bf16)e0[2 * v + 1]; pf.u[v]     = p0.u;
      PackU p1; p1.e[0] = (bf16)e1[2 * v]; p1.e[1] = (bf16)e1[2 * v + 1]; pf.u[4 + v] = p1.u;
    }

    // rescale O accumulators only while the running max is still moving
    // (uniform vote keeps EXEC full for the WMMAs below)
    if (__any(alpha != 1.0f)) {
      float af[8];
#pragma unroll
      for (int r = 0; r < 8; ++r) af[r] = lane_read(alpha, r + 8 * hh);
#pragma unroll
      for (int dt = 0; dt < 4; ++dt)
#pragma unroll
        for (int r = 0; r < 8; ++r) o[dt][r] *= af[r];
    }

    // O += P * V
    o[0] = wmma_bf16(pf.v, vb0, o[0]);
    o[1] = wmma_bf16(pf.v, vb1, o[1]);
    o[2] = wmma_bf16(pf.v, vb2, o[2]);
    o[3] = wmma_bf16(pf.v, vb3, o[3]);
    buf ^= 1;
  }

  // total softmax denominator per query (lane pair holds disjoint partial sums)
  float lt = l_run + lane_read(l_run, lane ^ 16);
  float inv[8];
#pragma unroll
  for (int r = 0; r < 8; ++r) inv[r] = 1.0f / lane_read(lt, r + 8 * hh);

  bf16* cbase = ctx + (size_t)(b * SEQ + q0) * HD + h * DH;
#pragma unroll
  for (int dt = 0; dt < 4; ++dt) {
    int col = dt * 16 + (lane & 15);
#pragma unroll
    for (int r = 0; r < 8; ++r) {
      int row = r + 8 * hh;
      cbase[(size_t)row * HD + col] = (bf16)(o[dt][r] * inv[r]);
    }
  }
}

extern "C" void kernel_launch(void* const* d_in, const int* in_sizes, int n_in,
                              void* d_out, int out_size, void* d_ws, size_t ws_size,
                              hipStream_t stream) {
  (void)in_sizes; (void)n_in; (void)out_size; (void)ws_size;
  const float* x  = (const float*)d_in[0];
  const float* Wq = (const float*)d_in[1];
  const float* bq = (const float*)d_in[2];
  const float* Wk = (const float*)d_in[3];
  const float* bk = (const float*)d_in[4];
  const float* Wv = (const float*)d_in[5];
  const float* bv = (const float*)d_in[6];
  const float* Wo = (const float*)d_in[7];
  const float* bo = (const float*)d_in[8];

  char* ws = (char*)d_ws;
  size_t off = 0;
  bf16* xb   = (bf16*)(ws + off); off += (size_t)MTOT * HD * 2;
  bf16* Wqb  = (bf16*)(ws + off); off += (size_t)HD * HD * 2;
  bf16* Wkb  = (bf16*)(ws + off); off += (size_t)HD * HD * 2;
  bf16* Wvb  = (bf16*)(ws + off); off += (size_t)HD * HD * 2;
  bf16* Wob  = (bf16*)(ws + off); off += (size_t)HD * HD * 2;
  bf16* qbuf = (bf16*)(ws + off); off += (size_t)MTOT * HD * 2;
  bf16* kbuf = (bf16*)(ws + off); off += (size_t)MTOT * HD * 2;
  bf16* vTb  = (bf16*)(ws + off); off += (size_t)MTOT * HD * 2;
  bf16* ctx  = (bf16*)(ws + off); off += (size_t)MTOT * HD * 2;

  const int nX4 = MTOT * HD / 4, nW4 = HD * HD / 4;
  cvt_f32_bf16<<<(nX4 + 255) / 256, 256, 0, stream>>>(x,  xb,  nX4);
  cvt_f32_bf16<<<(nW4 + 255) / 256, 256, 0, stream>>>(Wq, Wqb, nW4);
  cvt_f32_bf16<<<(nW4 + 255) / 256, 256, 0, stream>>>(Wk, Wkb, nW4);
  cvt_f32_bf16<<<(nW4 + 255) / 256, 256, 0, stream>>>(Wv, Wvb, nW4);
  cvt_f32_bf16<<<(nW4 + 255) / 256, 256, 0, stream>>>(Wo, Wob, nW4);

  dim3 g(MTOT / 128, HD / 128);
  gemm_bf16_wmma<0><<<g, 256, 0, stream>>>(xb, Wqb, bq, (void*)qbuf);
  gemm_bf16_wmma<0><<<g, 256, 0, stream>>>(xb, Wkb, bk, (void*)kbuf);
  gemm_bf16_wmma<1><<<g, 256, 0, stream>>>(xb, Wvb, bv, (void*)vTb);

  // 512 blocks: (b, h, 128-query slab) x 8 waves
  attention_wmma<<<BATCH * NH * (SEQ / 128), 256, 0, stream>>>(qbuf, kbuf, vTb, ctx);

  gemm_bf16_wmma<2><<<g, 256, 0, stream>>>(ctx, Wob, bo, d_out);
}
